// DCNGate_38517266710921
// MI455X (gfx1250) — matrix operand order repaired and checked by reading, not compiled, not run on hardware
//
#include <hip/hip_runtime.h>

typedef __attribute__((ext_vector_type(16))) _Float16 v16h;
typedef __attribute__((ext_vector_type(8)))  _Float16 v8h;
typedef __attribute__((ext_vector_type(8)))  float    v8f;

constexpr int BB = 4, C = 64, H = 192, W = 192;
constexpr int HW = H * W;
#define LN_EPS 1e-5f

// ---------------------------------------------------------------------------
// WMMA helper: D = A(16x32 f16) * B(32x16 f16) + C(16x16 f32), wave32.
// ---------------------------------------------------------------------------
__device__ __forceinline__ v8f wmma32(v16h a, v16h b, v8f c) {
  return __builtin_amdgcn_wmma_f32_16x16x32_f16(
      false, a, false, b, (short)0, c, false, false);
}

// A-fragment (16x32 f16) K index for element e of a lane.
// lanes 0-15 (kbase8=0): K=e
// lanes16-31 (kbase8=8): K=16+e
__device__ __forceinline__ int afrag_k(int kbase8, int e) {
  return 2 * kbase8 + e;
}
// B-fragment (32x16 f16): lanes 0-15 hold K=0..15, lanes 16-31 K=16..31 in
// element order; lane's column (pixel) = lane & 15.  With pixel-major f16
// activations this is ONE v16h (32B) load per lane per chunk.

__device__ __forceinline__ v16h ldv16h(const _Float16* p) {
  return *(const v16h*)p;
}

__device__ __forceinline__ v16h splat16(float s) {
  _Float16 h = (_Float16)s;
  v16h r;
#pragma unroll
  for (int i = 0; i < 16; ++i) r[i] = h;
  return r;
}

__device__ __forceinline__ void tile_decode(int t, int& b, int& y, int& x0) {
  constexpr int tilesPerRow = W / 16;
  constexpr int tilesPerImg = HW / 16;
  b = t / tilesPerImg;
  int r = t - b * tilesPerImg;
  y = r / tilesPerRow;
  x0 = (r - y * tilesPerRow) * 16;
}

// ---------------------------------------------------------------------------
// Weight pre-pack: f32 OIHW -> f16 blocks in exact A-fragment lane order.
// Block id = (tap*OCT + ot)*2 + ch ; each block = 32 lanes x 16 halves (1KB).
// Rows >= OC are zero-filled so padded tiles contribute nothing.
// ---------------------------------------------------------------------------
template <int KS, int OC, int OCT>
__global__ __launch_bounds__(32)
void pack_weights_kernel(const float* __restrict__ w, _Float16* __restrict__ wp) {
  int lane = threadIdx.x;
  int blk = blockIdx.x;                 // (tap*OCT + ot)*2 + ch
  int ch = blk & 1;
  int rest = blk >> 1;
  int ot = rest % OCT;
  int tap = rest / OCT;
  int ki = tap / KS, kj = tap - ki * KS;
  int kb8 = (lane < 16) ? 0 : 8;
  int o = ot * 16 + (lane & 15);
  v16h af;
#pragma unroll
  for (int e = 0; e < 16; ++e) {
    int c = ch * 32 + afrag_k(kb8, e);
    int osafe = (o < OC) ? o : (OC - 1);
    float wv = (o < OC) ? w[((osafe * C + c) * KS + ki) * KS + kj] : 0.f;
    af[e] = (_Float16)wv;
  }
  *(v16h*)(wp + (size_t)blk * 512 + lane * 16) = af;
}

// ---------------------------------------------------------------------------
// Kernel 1: LayerNorm(channel) + 1x1 conv (128x64) -> x1h, x2h (f16 pixel-major)
// ---------------------------------------------------------------------------
__global__ __launch_bounds__(128)
void ln_conv_in_kernel(const float* __restrict__ x,
                       const float* __restrict__ g, const float* __restrict__ beta,
                       const _Float16* __restrict__ wp, const float* __restrict__ b_in,
                       _Float16* __restrict__ x1h, _Float16* __restrict__ x2h) {
  int lane = threadIdx.x & 31;
  int t = blockIdx.x * 4 + (threadIdx.x >> 5);
  int b, y, x0;
  tile_decode(t, b, y, x0);
  int px = x0 + (lane & 15);
  int pix = b * HW + y * W + px;
  const float* xp = x + (size_t)b * C * HW + y * W + px;

  float mu = 0.f, s2 = 0.f;
#pragma unroll 8
  for (int c = 0; c < C; ++c) {
    float v = xp[(size_t)c * HW];
    mu += v; s2 += v * v;
  }
  mu *= (1.f / 64.f);
  float var = s2 * (1.f / 64.f) - mu * mu;
  float rs = __frsqrt_rn(var + LN_EPS);

  int kb16 = (lane < 16) ? 0 : 16;
  int base8 = (lane < 16) ? 0 : 8;

  v16h bf[2];
#pragma unroll
  for (int ch = 0; ch < 2; ++ch) {
#pragma unroll
    for (int e = 0; e < 16; ++e) {
      int c = ch * 32 + kb16 + e;
      float v = (xp[(size_t)c * HW] - mu) * rs * g[c] + beta[c];
      bf[ch][e] = (_Float16)v;
    }
  }

#pragma unroll 1
  for (int ot = 0; ot < 8; ++ot) {
    v8f acc = {};
#pragma unroll
    for (int ch = 0; ch < 2; ++ch) {
      v16h af = ldv16h(wp + (size_t)(ot * 2 + ch) * 512 + lane * 16);
      acc = wmma32(af, bf[ch], acc);
    }
    int o0 = ot * 16 + base8;                 // 8 consecutive out channels
    v8h hv;
#pragma unroll
    for (int e = 0; e < 8; ++e) hv[e] = (_Float16)(acc[e] + b_in[o0 + e]);
    if (o0 < 64) *(v8h*)(x1h + (size_t)pix * 64 + o0) = hv;
    else         *(v8h*)(x2h + (size_t)pix * 64 + (o0 - 64)) = hv;
  }
}

// ---------------------------------------------------------------------------
// Kernel 2: offset-generator conv2d (zero pad) -> off, f32 pixel-major,
// padded channel stride OCP = OCT*16.
// ---------------------------------------------------------------------------
template <int KS, int PAD, int OC, int OCT>
__global__ __launch_bounds__(128)
void offset_conv_kernel(const _Float16* __restrict__ inh,
                        const _Float16* __restrict__ wp,
                        const float* __restrict__ bias,
                        float* __restrict__ off) {
  constexpr int OCP = OCT * 16;
  int lane = threadIdx.x & 31;
  int t = blockIdx.x * 4 + (threadIdx.x >> 5);
  int b, y, x0;
  tile_decode(t, b, y, x0);
  int px = x0 + (lane & 15);
  int kb16 = (lane < 16) ? 0 : 16;
  int base8 = (lane < 16) ? 0 : 8;

  v8f zero = {};
  v8f acc[OCT];
#pragma unroll
  for (int i = 0; i < OCT; ++i) acc[i] = zero;

#pragma unroll 1
  for (int tap = 0; tap < KS * KS; ++tap) {
    int kh = tap / KS, kw = tap - kh * KS;
    int iy = y + kh - PAD;
    int ix = px + kw - PAD;
    float m = ((iy >= 0) && (iy < H) && (ix >= 0) && (ix < W)) ? 1.f : 0.f;
    int cy = min(max(iy, 0), H - 1);
    int cx = min(max(ix, 0), W - 1);
    int spix = b * HW + cy * W + cx;
    v16h mv = splat16(m);

    v16h bf[2];
#pragma unroll
    for (int ch = 0; ch < 2; ++ch)
      bf[ch] = ldv16h(inh + (size_t)spix * 64 + ch * 32 + kb16) * mv;

#pragma unroll
    for (int ot = 0; ot < OCT; ++ot) {
#pragma unroll
      for (int ch = 0; ch < 2; ++ch) {
        v16h af = ldv16h(wp + (size_t)((tap * OCT + ot) * 2 + ch) * 512 + lane * 16);
        acc[ot] = wmma32(af, bf[ch], acc[ot]);
      }
    }
  }
  int pixo = b * HW + y * W + px;
#pragma unroll
  for (int ot = 0; ot < OCT; ++ot) {
    int o0 = ot * 16 + base8;
    v8f ov;
#pragma unroll
    for (int e = 0; e < 8; ++e) {
      int o = o0 + e;
      ov[e] = acc[ot][e] + ((o < OC) ? bias[o] : 0.f);
    }
    *(v8f*)(off + (size_t)pixo * OCP + o0) = ov;
  }
}

// ---------------------------------------------------------------------------
// Kernel 3: deformable conv. Per tap: float2 offset read, 4-corner v16h
// gather with mask-folded bilinear weights (packed f16 FMA), 8 WMMAs.
// ---------------------------------------------------------------------------
template <int KS, int PAD, int OCP>
__global__ __launch_bounds__(128)
void deform_conv_kernel(const _Float16* __restrict__ inh,
                        const float* __restrict__ off,
                        const _Float16* __restrict__ wp,
                        const float* __restrict__ bias,
                        _Float16* __restrict__ outh) {
  int lane = threadIdx.x & 31;
  int t = blockIdx.x * 4 + (threadIdx.x >> 5);
  int b, y, x0;
  tile_decode(t, b, y, x0);
  int px = x0 + (lane & 15);
  int kb16 = (lane < 16) ? 0 : 16;
  int base8 = (lane < 16) ? 0 : 8;
  int pixo = b * HW + y * W + px;

  v8f zero = {};
  v8f acc[4];
#pragma unroll
  for (int i = 0; i < 4; ++i) acc[i] = zero;

#pragma unroll 1
  for (int k = 0; k < KS * KS; ++k) {
    int ki = k / KS, kj = k - ki * KS;
    float2 o2 = *(const float2*)(off + (size_t)pixo * OCP + 2 * k);
    float py  = (float)(y + ki - PAD) + o2.x;
    float pxf = (float)(px + kj - PAD) + o2.y;
    float fy = floorf(py), fx = floorf(pxf);
    float wy = py - fy, wx = pxf - fx;
    int y0 = (int)fy, xq = (int)fx;
    int y1 = y0 + 1, x1i = xq + 1;

    float m00 = ((y0 >= 0) && (y0 < H) && (xq  >= 0) && (xq  < W)) ? 1.f : 0.f;
    float m01 = ((y0 >= 0) && (y0 < H) && (x1i >= 0) && (x1i < W)) ? 1.f : 0.f;
    float m10 = ((y1 >= 0) && (y1 < H) && (xq  >= 0) && (xq  < W)) ? 1.f : 0.f;
    float m11 = ((y1 >= 0) && (y1 < H) && (x1i >= 0) && (x1i < W)) ? 1.f : 0.f;
    int cy0 = min(max(y0, 0), H - 1), cy1 = min(max(y1, 0), H - 1);
    int cx0 = min(max(xq, 0), W - 1), cx1 = min(max(x1i, 0), W - 1);
    int p00 = b * HW + cy0 * W + cx0, p01 = b * HW + cy0 * W + cx1;
    int p10 = b * HW + cy1 * W + cx0, p11 = b * HW + cy1 * W + cx1;
    v16h w00v = splat16((1.f - wy) * (1.f - wx) * m00);
    v16h w01v = splat16((1.f - wy) * wx * m01);
    v16h w10v = splat16(wy * (1.f - wx) * m10);
    v16h w11v = splat16(wy * wx * m11);

    v16h bf[2];
#pragma unroll
    for (int ch = 0; ch < 2; ++ch) {
      const _Float16* pc = inh + ch * 32 + kb16;
      v16h v = ldv16h(pc + (size_t)p00 * 64) * w00v;
      v = v + ldv16h(pc + (size_t)p01 * 64) * w01v;
      v = v + ldv16h(pc + (size_t)p10 * 64) * w10v;
      v = v + ldv16h(pc + (size_t)p11 * 64) * w11v;
      bf[ch] = v;
    }
#pragma unroll
    for (int ot = 0; ot < 4; ++ot) {
#pragma unroll
      for (int ch = 0; ch < 2; ++ch) {
        v16h af = ldv16h(wp + (size_t)((k * 4 + ot) * 2 + ch) * 512 + lane * 16);
        acc[ot] = wmma32(af, bf[ch], acc[ot]);
      }
    }
  }
#pragma unroll
  for (int ot = 0; ot < 4; ++ot) {
    int o0 = ot * 16 + base8;
    v8h hv;
#pragma unroll
    for (int e = 0; e < 8; ++e) hv[e] = (_Float16)(acc[ot][e] + bias[o0 + e]);
    *(v8h*)(outh + (size_t)pixo * 64 + o0) = hv;
  }
}

// ---------------------------------------------------------------------------
// Kernel 4: h = x1 + x2 (packed f16 add); out = w_out @ h + b_out + x (NCHW f32)
// ---------------------------------------------------------------------------
__global__ __launch_bounds__(128)
void add_conv_out_kernel(const _Float16* __restrict__ x1h,
                         const _Float16* __restrict__ x2h,
                         const float* __restrict__ xorig,
                         const _Float16* __restrict__ wp,
                         const float* __restrict__ b_out,
                         float* __restrict__ out) {
  int lane = threadIdx.x & 31;
  int t = blockIdx.x * 4 + (threadIdx.x >> 5);
  int b, y, x0;
  tile_decode(t, b, y, x0);
  int px = x0 + (lane & 15);
  int kb16 = (lane < 16) ? 0 : 16;
  int base8 = (lane < 16) ? 0 : 8;
  int pix = b * HW + y * W + px;

  v16h bf[2];
#pragma unroll
  for (int ch = 0; ch < 2; ++ch) {
    size_t idx = (size_t)pix * 64 + ch * 32 + kb16;
    bf[ch] = ldv16h(x1h + idx) + ldv16h(x2h + idx);
  }
#pragma unroll 1
  for (int ot = 0; ot < 4; ++ot) {
    v8f acc = {};
#pragma unroll
    for (int ch = 0; ch < 2; ++ch) {
      v16h af = ldv16h(wp + (size_t)(ot * 2 + ch) * 512 + lane * 16);
      acc = wmma32(af, bf[ch], acc);
    }
#pragma unroll
    for (int e = 0; e < 8; ++e) {
      int o = ot * 16 + base8 + e;
      size_t idx = (size_t)(b * 64 + o) * HW + y * W + px;
      out[idx] = acc[e] + b_out[o] + xorig[idx];
    }
  }
}

// ---------------------------------------------------------------------------
// Launch: weight pre-pack + full pipeline, workspace ping-pong in d_ws.
// ---------------------------------------------------------------------------
extern "C" void kernel_launch(void* const* d_in, const int* in_sizes, int n_in,
                              void* d_out, int out_size, void* d_ws, size_t ws_size,
                              hipStream_t stream) {
  const float* x     = (const float*)d_in[0];
  const float* ln_w  = (const float*)d_in[1];
  const float* ln_b  = (const float*)d_in[2];
  const float* w_in  = (const float*)d_in[3];
  const float* b_in  = (const float*)d_in[4];
  const float* w_out = (const float*)d_in[5];
  const float* b_out = (const float*)d_in[6];
  const float* dw1 = (const float*)d_in[7];
  const float* db1 = (const float*)d_in[8];
  const float* dw2 = (const float*)d_in[9];
  const float* db2 = (const float*)d_in[10];
  const float* dw3 = (const float*)d_in[11];
  const float* db3 = (const float*)d_in[12];
  const float* ow1 = (const float*)d_in[13];
  const float* ob1 = (const float*)d_in[14];
  const float* ow2 = (const float*)d_in[15];
  const float* ob2 = (const float*)d_in[16];
  const float* ow3 = (const float*)d_in[17];
  const float* ob3 = (const float*)d_in[18];

  const size_t NPIX = (size_t)BB * HW;      // 147456 pixels
  char* base = (char*)d_ws;
  float*     off  = (float*)base;                        base += NPIX * 112 * sizeof(float);
  _Float16*  x1a  = (_Float16*)base;                     base += NPIX * 64 * sizeof(_Float16);
  _Float16*  x1b  = (_Float16*)base;                     base += NPIX * 64 * sizeof(_Float16);
  _Float16*  x2h  = (_Float16*)base;                     base += NPIX * 64 * sizeof(_Float16);
  _Float16*  wpin  = (_Float16*)base;                    base += (size_t)16  * 512 * 2;
  _Float16*  wpout = (_Float16*)base;                    base += (size_t)8   * 512 * 2;
  _Float16*  wpo1  = (_Float16*)base;                    base += (size_t)350 * 512 * 2;
  _Float16*  wpo2  = (_Float16*)base;                    base += (size_t)200 * 512 * 2;
  _Float16*  wpo3  = (_Float16*)base;                    base += (size_t)36  * 512 * 2;
  _Float16*  wpd1  = (_Float16*)base;                    base += (size_t)392 * 512 * 2;
  _Float16*  wpd2  = (_Float16*)base;                    base += (size_t)200 * 512 * 2;
  _Float16*  wpd3  = (_Float16*)base;                    base += (size_t)72  * 512 * 2;

  dim3 b32(32);
  pack_weights_kernel<1, 128, 8><<<16,  b32, 0, stream>>>(w_in,  wpin);
  pack_weights_kernel<1, 64,  4><<<8,   b32, 0, stream>>>(w_out, wpout);
  pack_weights_kernel<5, 98,  7><<<350, b32, 0, stream>>>(ow1, wpo1);
  pack_weights_kernel<5, 50,  4><<<200, b32, 0, stream>>>(ow2, wpo2);
  pack_weights_kernel<3, 18,  2><<<36,  b32, 0, stream>>>(ow3, wpo3);
  pack_weights_kernel<7, 64,  4><<<392, b32, 0, stream>>>(dw1, wpd1);
  pack_weights_kernel<5, 64,  4><<<200, b32, 0, stream>>>(dw2, wpd2);
  pack_weights_kernel<3, 64,  4><<<72,  b32, 0, stream>>>(dw3, wpd3);

  int blocks = (int)(NPIX / 16 / 4);        // 4 waves x 16-pixel tiles
  dim3 blk(128);

  ln_conv_in_kernel<<<blocks, blk, 0, stream>>>(x, ln_w, ln_b, wpin, b_in, x1a, x2h);
  // stage 1: 5x5 offset conv (98ch, OCP=112) + 7x7 deform
  offset_conv_kernel<5, 2, 98, 7><<<blocks, blk, 0, stream>>>(x1a, wpo1, ob1, off);
  deform_conv_kernel<7, 3, 112><<<blocks, blk, 0, stream>>>(x1a, off, wpd1, db1, x1b);
  // stage 2: 5x5 offset conv (50ch, OCP=64) + 5x5 deform
  offset_conv_kernel<5, 2, 50, 4><<<blocks, blk, 0, stream>>>(x1b, wpo2, ob2, off);
  deform_conv_kernel<5, 2, 64><<<blocks, blk, 0, stream>>>(x1b, off, wpd2, db2, x1a);
  // stage 3: 3x3 offset conv (18ch, OCP=32) + 3x3 deform
  offset_conv_kernel<3, 1, 18, 2><<<blocks, blk, 0, stream>>>(x1a, wpo3, ob3, off);
  deform_conv_kernel<3, 1, 32><<<blocks, blk, 0, stream>>>(x1a, off, wpd3, db3, x1b);
  // merge + output projection + residual (NCHW f32 output)
  add_conv_out_kernel<<<blocks, blk, 0, stream>>>(x1b, x2h, x, wpout, b_out,
                                                  (float*)d_out);
}
